// DySample3d_32899449487703
// MI455X (gfx1250) — matrix operand order repaired and checked by reading, not compiled, not run on hardware
//
#include <hip/hip_runtime.h>
#include <hip/hip_bf16.h>
#include <math.h>

typedef __attribute__((ext_vector_type(16))) _Float16 v16h;
typedef __attribute__((ext_vector_type(8)))  float    v8f;
typedef __attribute__((ext_vector_type(4)))  float    v4f;

#define CD   64                 // channels
#define DD   32
#define HH   64
#define WW   64
#define DHW  (DD*HH*WW)         // 131072
#define DU   (DD*2)
#define HU   (HH*2)
#define WU   (WW*2)

// ---------------------------------------------------------------------------
// Kernel 0: pack w_off (96x64) and w_scope (96x64) into f16 [192][64].
// Row n = output channel N, K contiguous -> matches per-lane WMMA B fragment.
// ---------------------------------------------------------------------------
__global__ __launch_bounds__(256) void pack_weights(
    const float* __restrict__ w_off, const float* __restrict__ w_scope,
    _Float16* __restrict__ wcat)
{
    int i = blockIdx.x * 256 + threadIdx.x;
    if (i < 96 * 64)            wcat[i] = (_Float16)w_off[i];
    else if (i < 192 * 64)      wcat[i] = (_Float16)w_scope[i - 96 * 64];
}

// ---------------------------------------------------------------------------
// Kernel 1: transpose x [64][DHW] -> x_t [DHW][64]  (channel-last, f32).
// LDS-tiled, pad 65 for conflict-free banks. x is read once -> NT loads;
// x_t (33.5MB) is re-read heavily by the gather stage -> default RT stores
// keep it L2-resident (192MB L2 easily holds it).
// ---------------------------------------------------------------------------
__global__ __launch_bounds__(256) void transpose_xt(
    const float* __restrict__ x, float* __restrict__ xt)
{
    __shared__ float tile[64 * 65];
    const int tid = threadIdx.x;
    const int pbase = blockIdx.x * 64;
#pragma unroll
    for (int i = 0; i < 16; ++i) {
        int c  = i * 4 + (tid >> 6);
        int pl = tid & 63;
        tile[pl * 65 + c] =
            __builtin_nontemporal_load(&x[(size_t)c * DHW + pbase + pl]);
    }
    __syncthreads();
#pragma unroll
    for (int i = 0; i < 16; ++i) {
        int pl = i * 4 + (tid >> 6);
        int c  = tid & 63;
        xt[(size_t)(pbase + pl) * 64 + c] = tile[pl * 65 + c]; // coalesced write
    }
}

// ---------------------------------------------------------------------------
// Kernel 2: fused  (WMMA offset GEMM in LDS) + (trilinear grid-sample).
// Block = 256 threads (8 wave32), owns 128 coarse positions
//   (one d-slice, two h-rows, full w). Offsets never touch HBM.
// Output (256MB, write-only stream) uses NT stores so it cannot evict the
// L2-resident x_t that the 8-corner gathers depend on.
// ---------------------------------------------------------------------------
__global__ __launch_bounds__(256) void dysample_fused(
    const float*    __restrict__ xt,      // [DHW][64] channel-last
    const _Float16* __restrict__ wcat,    // [192][64] f16
    const float*    __restrict__ b_off,   // [96]
    const float*    __restrict__ b_scope, // [96]
    float*          __restrict__ out)     // [64][DU][HU][WU]
{
    __shared__ float offs[128][96];       // fused offsets, 48KB

    const int tid  = threadIdx.x;
    const int lane = tid & 31;
    const int wave = tid >> 5;
    const int p0   = blockIdx.x * 128;    // coarse linear base

    // ================= Stage 1: WMMA GEMM (off & scope, K=64) =============
    const int nrow = lane & 15;           // N (and M-row) index within tile
    const int hi   = lane >> 4;           // lane half
    const int p    = p0 + wave * 16 + nrow;

    // A fragments (16x32 f16), contiguous K:
    //  lane<16 : K {0..15};  lane>=16 : K {16..31}
    v16h a0, a1;
    {
        const float* ap = xt + (size_t)p * 64 + hi * 16;
#pragma unroll
        for (int j = 0; j < 8; ++j) {
            a0[j]     = (_Float16)ap[j];
            a0[j + 8] = (_Float16)ap[8 + j];
            a1[j]     = (_Float16)ap[32 + j];
            a1[j + 8] = (_Float16)ap[40 + j];
        }
    }

#pragma unroll
    for (int t = 0; t < 6; ++t) {         // 6 N-tiles cover 96 channels
        const int n = 16 * t + nrow;
        // B fragment: lane holds column N=n, 16 contiguous K halves
        const _Float16* bo = wcat + n * 64 + hi * 16;
        const _Float16* bs = wcat + (96 + n) * 64 + hi * 16;
        v16h b0 = *(const v16h*)(bo);
        v16h b1 = *(const v16h*)(bo + 32);
        v16h s0 = *(const v16h*)(bs);
        v16h s1 = *(const v16h*)(bs + 32);

        v8f co = {}, cs = {};
        co = __builtin_amdgcn_wmma_f32_16x16x32_f16(false, a0, false, b0, (short)0, co, false, false);
        co = __builtin_amdgcn_wmma_f32_16x16x32_f16(false, a1, false, b1, (short)0, co, false, false);
        cs = __builtin_amdgcn_wmma_f32_16x16x32_f16(false, a0, false, s0, (short)0, cs, false, false);
        cs = __builtin_amdgcn_wmma_f32_16x16x32_f16(false, a1, false, s1, (short)0, cs, false, false);

        const int   chan = 16 * t + nrow; // C/D layout: N = lane%16
        const float bofs = b_off[chan];
        const float bsc  = b_scope[chan];
#pragma unroll
        for (int r = 0; r < 8; ++r) {     // C/D layout: M = r + 8*(lane>=16)
            const int m = r + hi * 8;
            float o = co[r] + bofs;
            float s = cs[r] + bsc;
            float sig = 1.0f / (1.0f + __expf(-s));
            offs[wave * 16 + m][chan] = o * sig * 0.5f;
        }
    }
    __syncthreads();

    // ================= Stage 2: trilinear grid-sample ======================
    // 4096 tasks = g(4) x du(2) x hu(4) x wu(128); warp spans 32 contiguous wu
    const int d  = p0 >> 12;              // p0 / (H*W)
    const int h0 = (p0 >> 6) & 63;        // first of two h rows

    for (int it = 0; it < 16; ++it) {
        const int idx    = wave * 16 + it;
        const int wu     = (idx & 3) * 32 + lane;
        const int hul    = (idx >> 2) & 3;
        const int dul    = (idx >> 4) & 1;
        const int g      = (idx >> 5) & 3;
        const int hloc   = hul >> 1, r2 = hul & 1;
        const int du     = 2 * d + dul;
        const int hu     = 2 * (h0 + hloc) + r2;
        const int r3     = wu & 1,  wx = wu >> 1;
        const int lrow   = hloc * 64 + wx;
        // pixel-shuffle channel: o = (g*3+comp)*8 + (r1*4 + r2*2 + r3)
        const int ob     = g * 24 + (dul * 4 + r2 * 2 + r3);

        const float ox = offs[lrow][ob];
        const float oy = offs[lrow][ob + 8];
        const float oz = offs[lrow][ob + 16];

        // norm folds away: voxel coords directly
        float fx = fminf(fmaxf((float)wu * (63.0f / 127.0f) + ox, 0.0f), 63.0f);
        float fy = fminf(fmaxf((float)hu * (63.0f / 127.0f) + oy, 0.0f), 63.0f);
        float fz = fminf(fmaxf((float)du * (31.0f / 63.0f)  + oz, 0.0f), 31.0f);

        float x0f = floorf(fx), y0f = floorf(fy), z0f = floorf(fz);
        int X0 = (int)x0f, Y0 = (int)y0f, Z0 = (int)z0f;
        int X1 = min(X0 + 1, 63), Y1 = min(Y0 + 1, 63), Z1 = min(Z0 + 1, 31);
        float wx1 = fx - x0f, wy1 = fy - y0f, wz1 = fz - z0f;
        float wx0 = 1.0f - wx1, wy0 = 1.0f - wy1, wz0 = 1.0f - wz1;

        v4f acc0 = {}, acc1 = {}, acc2 = {}, acc3 = {};
        const float* basep = xt + g * 16;  // 16 contiguous channels per voxel
#pragma unroll
        for (int cz = 0; cz < 2; ++cz) {
            const int   Z   = cz ? Z1 : Z0;
            const float wz_ = cz ? wz1 : wz0;
#pragma unroll
            for (int cy = 0; cy < 2; ++cy) {
                const int   Y   = cy ? Y1 : Y0;
                const float wzy = wz_ * (cy ? wy1 : wy0);
#pragma unroll
                for (int cx = 0; cx < 2; ++cx) {
                    const int   X   = cx ? X1 : X0;
                    const float wgt = wzy * (cx ? wx1 : wx0);
                    const v4f* vp = (const v4f*)(basep +
                        (size_t)(((Z * 64 + Y) * 64 + X)) * 64);
                    acc0 += wgt * vp[0];
                    acc1 += wgt * vp[1];
                    acc2 += wgt * vp[2];
                    acc3 += wgt * vp[3];
                }
            }
        }

        // output [c][du][hu][wu]; lanes span wu -> 128B coalesced per channel.
        // NT stores: pure write stream, keep L2 for x_t.
        const size_t outbase =
            ((((size_t)(g * 16) * DU + du) * HU + hu) * WU) + wu;
        const size_t cstride = (size_t)DU * HU * WU;
#pragma unroll
        for (int c = 0; c < 16; ++c) {
            float v = (c < 4) ? acc0[c] : (c < 8) ? acc1[c - 4]
                    : (c < 12) ? acc2[c - 8] : acc3[c - 12];
            __builtin_nontemporal_store(v, &out[outbase + (size_t)c * cstride]);
        }
    }
}

// ---------------------------------------------------------------------------
extern "C" void kernel_launch(void* const* d_in, const int* in_sizes, int n_in,
                              void* d_out, int out_size, void* d_ws, size_t ws_size,
                              hipStream_t stream)
{
    const float* x       = (const float*)d_in[0];
    const float* w_off   = (const float*)d_in[1];
    const float* b_off   = (const float*)d_in[2];
    const float* w_scope = (const float*)d_in[3];
    const float* b_scope = (const float*)d_in[4];

    float*     xt   = (float*)d_ws;                               // 33,554,432 B
    _Float16*  wcat = (_Float16*)((char*)d_ws + (size_t)DHW * 64 * 4); // 24,576 B

    pack_weights <<<48,   256, 0, stream>>>(w_off, w_scope, wcat);
    transpose_xt <<<DHW / 64, 256, 0, stream>>>(x, xt);
    dysample_fused<<<DHW / 128, 256, 0, stream>>>(xt, wcat, b_off, b_scope,
                                                  (float*)d_out);
}
